// SpatialDecoder_85083302134341
// MI455X (gfx1250) — compile-verified, matrix operand order checked by reading, not run on hardware
//
#include <hip/hip_runtime.h>
#include <hip/hip_bf16.h>

typedef __attribute__((ext_vector_type(16))) __bf16 v16bf;
typedef __attribute__((ext_vector_type(8)))  __bf16 v8bf;
typedef __attribute__((ext_vector_type(8)))  float  v8f;
typedef __attribute__((ext_vector_type(4)))  unsigned int v4u;
typedef __attribute__((ext_vector_type(8)))  int v8i;
typedef __attribute__((ext_vector_type(4)))  int v4i;

#define HDIM 256
#define HAVE_TDM __has_builtin(__builtin_amdgcn_tensor_load_to_lds)

// ---------- helpers ----------
static __device__ inline unsigned short f2bf_bits(float f) {
    union { float f; unsigned u; } v; v.f = f;
    unsigned r = v.u + 0x7FFFu + ((v.u >> 16) & 1u);   // round-to-nearest-even
    return (unsigned short)(r >> 16);
}
static __device__ inline __bf16 bits2bf(unsigned short h) {
    __bf16 b; __builtin_memcpy(&b, &h, 2); return b;
}
static __device__ inline __bf16 f2bf(float f) {
    return bits2bf(f2bf_bits(f));
}
static __device__ inline float bf2f(__bf16 b) {
    unsigned short h; __builtin_memcpy(&h, &b, 2);
    union { unsigned u; float f; } v; v.u = ((unsigned)h) << 16; return v.f;
}
// order-preserving float<->uint mapping for atomicMax-based segment max
static __device__ inline unsigned fenc(float f) {
    unsigned u = __float_as_uint(f);
    return (u & 0x80000000u) ? ~u : (u | 0x80000000u);
}
static __device__ inline float fdec(unsigned e) {
    unsigned u = (e & 0x80000000u) ? (e & 0x7FFFFFFFu) : ~e;
    return __uint_as_float(u);
}

#if HAVE_TDM
// TDM: async-DMA a 2D bf16 tile (64 cols x 256 rows, row stride 256 elems) into LDS.
// D# built per CDNA5 ISA ch.8: group0 = {count/flags, lds_addr, global_addr, type=2},
// group1 = {data_size=2B, tensor/tile dims, strides}, groups 2/3 zero (2-D tensor).
static __device__ inline void tdm_load_panel_2d(unsigned lds_off, const void* gsrc) {
    unsigned long long ga = (unsigned long long)(uintptr_t)gsrc;
    v4u g0;
    g0[0] = 1u;                                             // count=1, user mode
    g0[1] = lds_off;                                        // lds_addr (bytes)
    g0[2] = (unsigned)ga;                                   // global_addr[31:0]
    g0[3] = (unsigned)((ga >> 32) & 0x01FFFFFFu) | (2u << 30); // addr[56:32] | type=2
    v8i g1;
    g1[0] = 0x00010000;        // workgroup_mask=0, data_size=1 (2 bytes)
    g1[1] = 64 << 16;          // tensor_dim0 = 64 (tile fully in-bounds)
    g1[2] = 256 << 16;         // tensor_dim0 hi=0 | tensor_dim1 = 256 (lo16)
    g1[3] = 64 << 16;          // tensor_dim1 hi=0 | tile_dim0 = 64
    g1[4] = 256;               // tile_dim1 = 256, tile_dim2 = 0
    g1[5] = 256;               // tensor_dim0_stride[31:0] = 256
    g1[6] = 0;                 // stride0 hi | stride1 lo
    g1[7] = 0;
    v4i gz = {0, 0, 0, 0};
#if defined(__clang_major__) && __clang_major__ >= 23
    v8i gz8 = {0, 0, 0, 0, 0, 0, 0, 0};
    __builtin_amdgcn_tensor_load_to_lds(g0, g1, gz, gz, gz8, 0);
#else
    __builtin_amdgcn_tensor_load_to_lds(g0, g1, gz, gz, 0);
#endif
}
#endif

// ---------- per-layer W -> bf16 hi/lo split (row-major [256][256]) ----------
__global__ void convert_w_kernel(const float* __restrict__ W,
                                 unsigned short* __restrict__ w_hi,
                                 unsigned short* __restrict__ w_lo, int n) {
    int i = blockIdx.x * 256 + threadIdx.x;
    if (i >= n) return;
    float w = W[i];
    unsigned short hb = f2bf_bits(w);
    w_hi[i] = hb;
    w_lo[i] = f2bf_bits(w - bf2f(bits2bf(hb)));
}

// ---------- GEMM: h = x @ W^T, fused s_i = h@a[:H], s_j = h@a[H:] ----------
// Block = 256 threads = 8 waves; wave w computes rows [blk*128 + 16w, +16) x all 256 cols.
// bf16x3 split-fp32 via v_wmma_f32_16x16x32_bf16. W panels (64-wide K) are DMA'd into a
// double-buffered LDS stage by the Tensor Data Mover, overlapped with WMMA compute.
__global__ __launch_bounds__(256) void gat_gemm_kernel(
    const float* __restrict__ x,                 // [NN,256]
    const unsigned short* __restrict__ w_hi_g,   // [256,256] bf16 bits
    const unsigned short* __restrict__ w_lo_g,   // [256,256] bf16 bits
    const float* __restrict__ a,                 // [512]
    float* __restrict__ h,                       // [NN,256]
    float* __restrict__ si,                      // [NN]
    float* __restrict__ sj)                      // [NN]
{
    extern __shared__ char smem[];
    __bf16* lds = (__bf16*)smem;   // 2 buffers x (hi[256][64] + lo[256][64]) = 128KB

    const int tid  = threadIdx.x;
    const int wave = tid >> 5;
    const int lane = tid & 31;
    const int half = (lane < 16) ? 0 : 1;    // which 16-lane half
    const int m0   = blockIdx.x * 128 + wave * 16;
    const int row  = m0 + (lane & 15);
    const float* xrow = x + (size_t)row * HDIM;

    v8f acc[16];
#pragma unroll
    for (int t = 0; t < 16; ++t)
#pragma unroll
        for (int e = 0; e < 8; ++e) acc[t][e] = 0.f;

    const int kbA = half ? 8 : 0;            // A-frag K base per half-wave
    const int kbB = half ? 8 : 0;            // B-frag mirrors A layout

#if HAVE_TDM
    if (wave == 0) {   // prologue: DMA panel 0 into buffer 0 (TDM ignores EXEC)
        tdm_load_panel_2d((unsigned)(uintptr_t)(lds),         w_hi_g);
        tdm_load_panel_2d((unsigned)(uintptr_t)(lds + 16384), w_lo_g);
    }
#endif

    for (int p = 0; p < 4; ++p) {            // 4 K-panels of 64
        const int kp = p * 64;
        __bf16* cb_hi = lds + (p & 1) * 32768;
        __bf16* cb_lo = cb_hi + 16384;
        __builtin_prefetch(xrow + ((kp + 64) & 255), 0, 3);
#if HAVE_TDM
        if (wave == 0) {
            if (p < 3) {
                __bf16* nb = lds + ((p + 1) & 1) * 32768;   // freed at end of iter p-1
                tdm_load_panel_2d((unsigned)(uintptr_t)(nb),         w_hi_g + kp + 64);
                tdm_load_panel_2d((unsigned)(uintptr_t)(nb + 16384), w_lo_g + kp + 64);
                __builtin_amdgcn_s_wait_tensorcnt(2);  // in-order: panel p done
            } else {
                __builtin_amdgcn_s_wait_tensorcnt(0);
            }
        }
        __syncthreads();                      // release waves onto panel p
#else
        __syncthreads();
        for (int i = tid; i < 16384; i += 256) {
            int n = i >> 6, kk = i & 63;
            cb_hi[i] = bits2bf(w_hi_g[n * HDIM + kp + kk]);
            cb_lo[i] = bits2bf(w_lo_g[n * HDIM + kp + kk]);
        }
        __syncthreads();
#endif

        for (int k0 = kp; k0 < kp + 64; k0 += 32) {
            // ---- A fragment (per ISA 16-bit A 16x32 layout) ----
            v16bf a_hi, a_lo;
#pragma unroll
            for (int e = 0; e < 8; ++e) {
                float v0 = xrow[k0 + kbA + e];        // K = kb + e       (VGPR 0-3)
                float v1 = xrow[k0 + kbA + 16 + e];   // K = kb + 16 + e  (VGPR 4-7)
                __bf16 h0 = f2bf(v0), h1 = f2bf(v1);
                a_hi[e]     = h0;  a_lo[e]     = f2bf(v0 - bf2f(h0));
                a_hi[e + 8] = h1;  a_lo[e + 8] = f2bf(v1 - bf2f(h1));
            }
            const int koff = k0 - kp;                 // 0 or 32 within panel
#pragma unroll
            for (int t = 0; t < 16; ++t) {
                int n = t * 16 + (lane & 15);         // B-frag column for this lane
                const __bf16* ph = cb_hi + n * 64 + koff + kbB;
                const __bf16* pl = cb_lo + n * 64 + koff + kbB;
                v8bf bh0 = *(const v8bf*)(ph);
                v8bf bh1 = *(const v8bf*)(ph + 16);
                v8bf bl0 = *(const v8bf*)(pl);
                v8bf bl1 = *(const v8bf*)(pl + 16);
                v16bf b_hi = __builtin_shufflevector(bh0, bh1, 0,1,2,3,4,5,6,7,8,9,10,11,12,13,14,15);
                v16bf b_lo = __builtin_shufflevector(bl0, bl1, 0,1,2,3,4,5,6,7,8,9,10,11,12,13,14,15);
                acc[t] = __builtin_amdgcn_wmma_f32_16x16x32_bf16(false, a_hi, false, b_hi, (short)0, acc[t], false, false);
                acc[t] = __builtin_amdgcn_wmma_f32_16x16x32_bf16(false, a_hi, false, b_lo, (short)0, acc[t], false, false);
                acc[t] = __builtin_amdgcn_wmma_f32_16x16x32_bf16(false, a_lo, false, b_hi, (short)0, acc[t], false, false);
            }
        }
        __syncthreads();                      // everyone done with buffer (p&1)
    }

    // ---- epilogue: store h, fused GEMVs s_i/s_j ----
    const int mrow = m0 + (half ? 8 : 0);    // C/D layout: VGPR r -> M = r (+8 for upper half)
    float pi[8], pj[8];
#pragma unroll
    for (int r = 0; r < 8; ++r) { pi[r] = 0.f; pj[r] = 0.f; }

#pragma unroll
    for (int t = 0; t < 16; ++t) {
        int n = t * 16 + (lane & 15);
        float av = a[n];
        float bv = a[HDIM + n];
#pragma unroll
        for (int r = 0; r < 8; ++r) {
            float v = acc[t][r];
            h[(size_t)(mrow + r) * HDIM + n] = v;
            pi[r] += v * av;
            pj[r] += v * bv;
        }
    }
    // reduce over the 16 lanes of each half (same row, different n)
#pragma unroll
    for (int r = 0; r < 8; ++r) {
        float vi = pi[r], vj = pj[r];
#pragma unroll
        for (int m = 8; m >= 1; m >>= 1) {
            vi += __shfl_xor(vi, m, 32);
            vj += __shfl_xor(vj, m, 32);
        }
        if ((lane & 15) == 0) {
            si[mrow + r] = vi;
            sj[mrow + r] = vj;
        }
    }
}

// ---------- utility fills ----------
__global__ void fill_f32_kernel(float* __restrict__ p, float v, int n) {
    int i = blockIdx.x * 256 + threadIdx.x;
    if (i < n) p[i] = v;
}
__global__ void fill_u32_kernel(unsigned* __restrict__ p, unsigned v, int n) {
    int i = blockIdx.x * 256 + threadIdx.x;
    if (i < n) p[i] = v;
}

// ---------- per-dst in-degree (layer-invariant) ----------
__global__ void gat_count_kernel(const int* __restrict__ dst, float* __restrict__ cnt, int E) {
    int e = blockIdx.x * 256 + threadIdx.x;
    if (e >= E) return;
    unsafeAtomicAdd(&cnt[dst[e]], 1.0f);
}

// ---------- edge score + segment max ----------
__global__ void gat_score_kernel(const int* __restrict__ src, const int* __restrict__ dst,
                                 const float* __restrict__ si, const float* __restrict__ sj,
                                 float* __restrict__ alpha, unsigned* __restrict__ amax, int E) {
    int e = blockIdx.x * 256 + threadIdx.x;
    if (e >= E) return;
    int d = dst[e], s = src[e];
    float sc = si[d] + sj[s];
    float al = (sc > 0.f) ? sc : 0.2f * sc;      // leaky_relu(0.2)
    alpha[e] = al;
    atomicMax(&amax[d], fenc(al));
}

// ---------- exp(a - amax[dst]) + segment sum ----------
__global__ void gat_expsum_kernel(const int* __restrict__ dst,
                                  float* __restrict__ alpha, const unsigned* __restrict__ amax,
                                  float* __restrict__ denom, int E) {
    int e = blockIdx.x * 256 + threadIdx.x;
    if (e >= E) return;
    int d = dst[e];
    float m = fdec(amax[d]);
    float ex = expf(alpha[e] - m);
    alpha[e] = ex;                                // reuse buffer for ex
    unsafeAtomicAdd(&denom[d], ex);
}

// ---------- message scatter: sums[dst] += h[src] * (ex/denom[dst]) ----------
// one wave per edge, 32 lanes x 8 channels = 256
__global__ __launch_bounds__(256) void gat_scatter_kernel(
    const int* __restrict__ src, const int* __restrict__ dst,
    const float* __restrict__ h, const float* __restrict__ ex,
    const float* __restrict__ denom, float* __restrict__ sums, int E) {
    int e = blockIdx.x * 8 + (threadIdx.x >> 5);
    if (e >= E) return;
    int lane = threadIdx.x & 31;
    int s = src[e], d = dst[e];
    float w = ex[e] / denom[d];
    const float* hs = h + (size_t)s * HDIM;
    float* sd = sums + (size_t)d * HDIM;
#pragma unroll
    for (int i = 0; i < 8; ++i) {
        int c = lane + 32 * i;
        unsafeAtomicAdd(&sd[c], hs[c] * w);
    }
}

// ---------- mean + ELU (in place capable) ----------
__global__ void gat_finalize_kernel(const float* __restrict__ sums, const float* __restrict__ cnt,
                                    float* __restrict__ xout, int total) {
    int i = blockIdx.x * 256 + threadIdx.x;
    if (i >= total) return;
    int node = i >> 8;                            // /256
    float c = cnt[node]; if (c < 1.f) c = 1.f;
    float v = sums[i] / c;
    xout[i] = (v > 0.f) ? v : expm1f(v);          // elu
}

extern "C" void kernel_launch(void* const* d_in, const int* in_sizes, int n_in,
                              void* d_out, int out_size, void* d_ws, size_t ws_size,
                              hipStream_t stream) {
    const float* x0 = (const float*)d_in[0];
    const int*   ei = (const int*)d_in[1];       // jax default x64-off -> int32
    const int NN = in_sizes[0] / HDIM;           // 32768
    const int E  = in_sizes[1] / 2;              // 262144
    const int HTOT = NN * HDIM;
    const int* srcp = ei;
    const int* dstp = ei + E;

    float*          bufA  = (float*)d_ws;        // x / sums ping buffer
    float*          bufH  = bufA + HTOT;         // h
    float*          si    = bufH + HTOT;
    float*          sj    = si + NN;
    unsigned*       amax  = (unsigned*)(sj + NN);
    float*          denom = (float*)(amax + NN);
    float*          cnt   = denom + NN;
    float*          alpha = cnt + NN;            // E floats
    unsigned short* whi   = (unsigned short*)(alpha + E);   // 256*256 bf16 bits
    unsigned short* wlo   = whi + HDIM * HDIM;

    dim3 blk(256);
    dim3 gE((E + 255) / 256);
    dim3 gN((NN + 255) / 256);
    dim3 gH((HTOT + 255) / 256);
    dim3 gW((HDIM * HDIM + 255) / 256);
    dim3 gGemm(NN / 128);
    dim3 gScat((E + 7) / 8);
    const int ldsBytes = 2 * 2 * 256 * 64 * (int)sizeof(__bf16);   // 128KB double-buffer
    hipFuncSetAttribute((const void*)gat_gemm_kernel,
                        hipFuncAttributeMaxDynamicSharedMemorySize, ldsBytes);

    // in-degree (dst-only, identical across layers)
    fill_f32_kernel<<<gN, blk, 0, stream>>>(cnt, 0.f, NN);
    gat_count_kernel<<<gE, blk, 0, stream>>>(dstp, cnt, E);

    for (int L = 0; L < 3; ++L) {
        const float* xin  = (L == 0) ? x0 : bufA;
        float*       sums = (L == 2) ? (float*)d_out : bufA;
        const float* W    = (const float*)d_in[2 + 2 * L];
        const float* a    = (const float*)d_in[3 + 2 * L];

        convert_w_kernel<<<gW, blk, 0, stream>>>(W, whi, wlo, HDIM * HDIM);
        gat_gemm_kernel<<<gGemm, blk, ldsBytes, stream>>>(xin, whi, wlo, a, bufH, si, sj);

        fill_u32_kernel<<<gN, blk, 0, stream>>>(amax, 0u, NN);
        fill_f32_kernel<<<gN, blk, 0, stream>>>(denom, 0.f, NN);
        fill_f32_kernel<<<gH, blk, 0, stream>>>(sums, 0.f, HTOT);

        gat_score_kernel <<<gE, blk, 0, stream>>>(srcp, dstp, si, sj, alpha, amax, E);
        gat_expsum_kernel<<<gE, blk, 0, stream>>>(dstp, alpha, amax, denom, E);
        gat_scatter_kernel<<<gScat, blk, 0, stream>>>(srcp, dstp, bufH, alpha, denom, sums, E);
        gat_finalize_kernel<<<gH, blk, 0, stream>>>(sums, cnt, sums, HTOT);
    }
}
